// TSPTopKModel_7069516169265
// MI455X (gfx1250) — matrix operand order repaired and checked by reading, not compiled, not run on hardware
//
#include <hip/hip_runtime.h>

// Problem constants
#define BB 64
#define PP 500
#define NN 500
#define EE 128
#define HH 8
#define DD 16
#define HD 128
#define NP 512   // P/N padded to multiple of 16

typedef __attribute__((ext_vector_type(16))) _Float16 v16h;
typedef __attribute__((ext_vector_type(8)))  _Float16 v8h;
typedef __attribute__((ext_vector_type(8)))  float    v8f;

static __device__ __forceinline__ v8f wmma16(v16h a, v16h b, v8f c) {
  // D = A(16x32 f16) * B(32x16 f16) + C(16x16 f32)
  return __builtin_amdgcn_wmma_f32_16x16x32_f16(false, a, false, b, (short)0, c,
                                                false, false);
}
static __device__ __forceinline__ v16h cat8(v8h lo, v8h hi) {
  return __builtin_shufflevector(lo, hi, 0, 1, 2, 3, 4, 5, 6, 7,
                                 8, 9, 10, 11, 12, 13, 14, 15);
}
static __device__ __forceinline__ v8h  ld8(const _Float16* p)  { return *(const v8h*)p; }
static __device__ __forceinline__ v16h ld16(const _Float16* p) { return *(const v16h*)p; }

// Cross-lane reductions over the 16-lane group sharing C-matrix rows
// (xor of lane bits 0..3 never crosses the half-wave boundary).
static __device__ __forceinline__ float red_max16(float v) {
  v = fmaxf(v, __int_as_float(__builtin_amdgcn_ds_swizzle(__float_as_int(v), 0x1f | (1 << 10))));
  v = fmaxf(v, __int_as_float(__builtin_amdgcn_ds_swizzle(__float_as_int(v), 0x1f | (2 << 10))));
  v = fmaxf(v, __int_as_float(__builtin_amdgcn_ds_swizzle(__float_as_int(v), 0x1f | (4 << 10))));
  v = fmaxf(v, __int_as_float(__builtin_amdgcn_ds_swizzle(__float_as_int(v), 0x1f | (8 << 10))));
  return v;
}
static __device__ __forceinline__ float red_sum16(float v) {
  v += __int_as_float(__builtin_amdgcn_ds_swizzle(__float_as_int(v), 0x1f | (1 << 10)));
  v += __int_as_float(__builtin_amdgcn_ds_swizzle(__float_as_int(v), 0x1f | (2 << 10)));
  v += __int_as_float(__builtin_amdgcn_ds_swizzle(__float_as_int(v), 0x1f | (4 << 10)));
  v += __int_as_float(__builtin_amdgcn_ds_swizzle(__float_as_int(v), 0x1f | (8 << 10)));
  return v;
}

// ---------------- prep kernels ----------------
// Weights converted to f16 AND pre-swizzled into WMMA B-fragment order:
// dst[(((m*8 + ct)*4 + kc)*32 + lane)*16 + e] = src[(kc*32 + 8*(lane>>4) + (e&7) + 16*(e>>3))*HD + ct*16 + (lane&15)]
// so each lane's 16-element fragment is one contiguous 32B block.
__global__ void prep_weights(const float* __restrict__ wqf, const float* __restrict__ wql,
                             const float* __restrict__ wk,  const float* __restrict__ wv,
                             const float* __restrict__ wcomb, _Float16* __restrict__ dst) {
  int i = blockIdx.x * blockDim.x + threadIdx.x;
  if (i >= 5 * EE * HD) return;
  int e    = i & 15;
  int lane = (i >> 4) & 31;
  int kc   = (i >> 9) & 3;
  int ct   = (i >> 11) & 7;
  int m    = i >> 14;
  const float* src = (m == 0) ? wqf : (m == 1) ? wql : (m == 2) ? wk : (m == 3) ? wv : wcomb;
  int K   = kc * 32 + 8 * (lane >> 4) + (e & 7) + 16 * (e >> 3);   // bmap
  int col = ct * 16 + (lane & 15);
  dst[i] = (_Float16)src[K * HD + col];
}

__global__ void prep_enc(const float* __restrict__ enc, _Float16* __restrict__ enc_h) {
  int i = blockIdx.x * blockDim.x + threadIdx.x;
  if (i >= BB * NP * EE) return;
  int e = i & (EE - 1);
  int n = (i >> 7) & (NP - 1);
  int b = i >> 16;          // NP*EE == 65536
  enc_h[i] = (n < NN) ? (_Float16)enc[((size_t)b * NN + n) * EE + e] : (_Float16)0.f;
}

// ---------------- QKV projection ----------------
// one wave per (b, 16-node tile): 16x128 strip of Q, K, V
// Q,K -> (B,H,NP,D) row-major (via LDS transpose, 32B row stores)
// V   -> (B,H,D,NP) transposed (direct 16B vector stores)
__global__ __launch_bounds__(32)
void qkv_kernel(const _Float16* __restrict__ enc_h,
                const _Float16* __restrict__ wqf, const _Float16* __restrict__ wql,
                const _Float16* __restrict__ wk,  const _Float16* __restrict__ wv,
                const int* __restrict__ cur,
                _Float16* __restrict__ qh, _Float16* __restrict__ kh,
                _Float16* __restrict__ vt) {
  __shared__ __align__(32) _Float16 sq[16][16];
  __shared__ __align__(32) _Float16 sk[16][16];
  int blk = blockIdx.x;
  int b = blk >> 5, tile = blk & 31;
  int lane = threadIdx.x, lm = lane & 15, hl = lane >> 4;
  int m0 = tile * 16;

  const _Float16* encb = enc_h + (size_t)b * NP * EE;
  int rowE = m0 + lm;
  int rowL = (rowE < PP) ? cur[b * PP + rowE] : 0;

  v16h aE[4], aL[4];
#pragma unroll
  for (int kc = 0; kc < 4; ++kc) {
    const _Float16* pe = encb + (size_t)rowE * EE + kc * 32 + 8 * hl;
    const _Float16* pl = encb + (size_t)rowL * EE + kc * 32 + 8 * hl;
    aE[kc] = cat8(ld8(pe), ld8(pe + 16));
    aL[kc] = cat8(ld8(pl), ld8(pl + 16));
  }

  for (int ct = 0; ct < 8; ++ct) {          // head ct
    v8f accK = {}, accV = {}, accQ = {};
#pragma unroll
    for (int kc = 0; kc < 4; ++kc) {
      size_t fo = (size_t)((ct * 4 + kc) * 32 + lane) * 16;
      v16h bk  = ld16(wk  + fo);
      v16h bv  = ld16(wv  + fo);
      v16h bqf = ld16(wqf + fo);
      v16h bql = ld16(wql + fo);
      accK = wmma16(aE[kc], bk,  accK);
      accV = wmma16(aE[kc], bv,  accV);
      accQ = wmma16(aE[kc], bqf, accQ);
      accQ = wmma16(aL[kc], bql, accQ);
    }
    // V transposed: element r -> node m0+8*hl+r, contiguous per lane
    v8h vv;
#pragma unroll
    for (int r = 0; r < 8; ++r) vv[r] = (_Float16)accV[r];
    *(v8h*)(vt + ((size_t)(b * HH + ct) * DD + lm) * NP + m0 + 8 * hl) = vv;

    // Q,K: LDS transpose then coalesced 32B row stores (q by lanes 0-15, k by 16-31)
    __syncthreads();
#pragma unroll
    for (int r = 0; r < 8; ++r) {
      sq[r + 8 * hl][lm] = (_Float16)accQ[r];
      sk[r + 8 * hl][lm] = (_Float16)accK[r];
    }
    __syncthreads();
    _Float16* dstp = (hl ? kh : qh) + ((size_t)(b * HH + ct) * NP + m0 + lm) * DD;
    const _Float16* srcp = hl ? &sk[lm][0] : &sq[lm][0];
    *(v16h*)dstp = ld16(srcp);
  }
}

// ---------------- attention ----------------
// one wave per (b, h, 16-row p tile); exact 2-pass softmax through LDS
__global__ __launch_bounds__(32)
void attn_kernel(const _Float16* __restrict__ qh, const _Float16* __restrict__ kh,
                 const _Float16* __restrict__ vt, const float* __restrict__ mask,
                 _Float16* __restrict__ attn_h) {
  __shared__ __align__(16) float    sc[16][NP];   // 32 KB scores
  __shared__ __align__(32) _Float16 wh[16][NP];   // 16 KB softmax weights (f16)
  __shared__ __align__(32) _Float16 st[16][16];   // output tile transpose
  int blk = blockIdx.x;
  int pt = blk & 31; blk >>= 5;
  int h = blk & 7;   int b = blk >> 3;
  int lane = threadIdx.x, lm = lane & 15, hl = lane >> 4;
  int m0 = pt * 16;

  const _Float16* qb = qh + (size_t)(b * HH + h) * NP * DD;
  const _Float16* kb = kh + (size_t)(b * HH + h) * NP * DD;
  const _Float16* vb = vt + (size_t)(b * HH + h) * DD * NP;

  // q A-fragment: K = d (0..15 real), K>=16 zero
  v8h z8 = {};
  v16h qa = cat8(ld8(qb + (size_t)(m0 + lm) * DD + 8 * hl), z8);

  float rmax[8];
#pragma unroll
  for (int r = 0; r < 8; ++r) rmax[r] = -1e30f;

  for (int nt = 0; nt < 32; ++nt) {
    int n = nt * 16 + lm;
    // B-fragment of k^T: elements 0-7 hold K=8*hl..8*hl+7 (=d), elements 8-15 are zero pad
    v16h kf = cat8(ld8(kb + (size_t)n * DD + 8 * hl), z8);
    v8f s = {};
    s = wmma16(qa, kf, s);
#pragma unroll
    for (int r = 0; r < 8; ++r) {
      int prow = m0 + r + 8 * hl;
      float val;
      if (n >= NN) {
        val = -1e30f;
      } else {
        val = s[r] * 0.25f;                 // 1/sqrt(D)
        if (prow < PP) val += mask[((size_t)b * PP + prow) * NN + n];
      }
      sc[r + 8 * hl][nt * 16 + lm] = val;
      rmax[r] = fmaxf(rmax[r], val);
    }
  }
#pragma unroll
  for (int r = 0; r < 8; ++r) rmax[r] = red_max16(rmax[r]);
  __syncthreads();

  float rsum[8];
#pragma unroll
  for (int r = 0; r < 8; ++r) rsum[r] = 0.f;
  for (int nt = 0; nt < 32; ++nt) {
#pragma unroll
    for (int r = 0; r < 8; ++r) {
      int row = r + 8 * hl;
      float e_ = __expf(sc[row][nt * 16 + lm] - rmax[r]);
      wh[row][nt * 16 + lm] = (_Float16)e_;
      rsum[r] += e_;
    }
  }
#pragma unroll
  for (int r = 0; r < 8; ++r) rsum[r] = red_sum16(rsum[r]);
  __syncthreads();                           // expose w grid for transposed reads

  // out = w @ v, K = 512 in 16 chunks of 32; w from LDS (ds_load_b128), v contiguous
  v8f acc = {};
  for (int kc = 0; kc < 16; ++kc) {
    const _Float16* wrow = &wh[lm][kc * 32 + 8 * hl];
    v16h wa = cat8(ld8(wrow), ld8(wrow + 16));
    v16h vf = cat8(ld8(vb + (size_t)lm * NP + kc * 32 + 8 * hl), ld8(vb + (size_t)lm * NP + kc * 32 + 8 * hl + 16));
    acc = wmma16(wa, vf, acc);
  }
  // normalize rows, LDS transpose, coalesced 16B stores
#pragma unroll
  for (int r = 0; r < 8; ++r)
    st[r + 8 * hl][lm] = (_Float16)(acc[r] * (1.0f / rsum[r]));
  __syncthreads();
  *(v8h*)(attn_h + ((size_t)b * NP + m0 + lm) * HD + h * 16 + 8 * hl) =
      ld8(&st[lm][8 * hl]);
}

// ---------------- combine + final pointer scores ----------------
__global__ __launch_bounds__(32)
void final_kernel(const _Float16* __restrict__ attn_h, const _Float16* __restrict__ wcomb,
                  const float* __restrict__ bcomb, const _Float16* __restrict__ enc_h,
                  const float* __restrict__ mask, float* __restrict__ out) {
  __shared__ __align__(16) float    ex[16][NP];   // 32 KB
  __shared__ __align__(32) _Float16 ch[16][EE];   // combined 16x128 tile
  __shared__ float rinv[16];
  int blk = blockIdx.x;
  int b = blk >> 5, pt = blk & 31;
  int lane = threadIdx.x, lm = lane & 15, hl = lane >> 4;
  int m0 = pt * 16;

  // A fragments from attention output (16 x 128)
  v16h aA[4];
#pragma unroll
  for (int kc = 0; kc < 4; ++kc) {
    const _Float16* pa = attn_h + ((size_t)b * NP + m0 + lm) * HD + kc * 32 + 8 * hl;
    aA[kc] = cat8(ld8(pa), ld8(pa + 16));
  }

  // comb = attn @ W_comb + b_comb  ->  LDS (row-major f16)
  for (int ct = 0; ct < 8; ++ct) {
    int col = ct * 16 + lm;
    v8f acc = {};
#pragma unroll
    for (int kc = 0; kc < 4; ++kc) {
      v16h bf = ld16(wcomb + (size_t)((ct * 4 + kc) * 32 + lane) * 16);
      acc = wmma16(aA[kc], bf, acc);
    }
    float bias = bcomb[col];
#pragma unroll
    for (int r = 0; r < 8; ++r) ch[r + 8 * hl][col] = (_Float16)(acc[r] + bias);
  }
  __syncthreads();

  // A fragments of comb (transposed read out of LDS, ds_load_b128)
  v16h cA[4];
#pragma unroll
  for (int kc = 0; kc < 4; ++kc) {
    const _Float16* pc = &ch[lm][kc * 32 + 8 * hl];
    cA[kc] = cat8(ld8(pc), ld8(pc + 16));
  }

  const float inv_sqrt_e = 0.08838834764831845f; // 1/sqrt(128)
  float rsum[8];
#pragma unroll
  for (int r = 0; r < 8; ++r) rsum[r] = 0.f;

  for (int nt = 0; nt < 32; ++nt) {
    int n = nt * 16 + lm;
    v8f acc = {};
#pragma unroll
    for (int kc = 0; kc < 4; ++kc) {
      v16h be = cat8(ld8(enc_h + ((size_t)b * NP + n) * EE + kc * 32 + 8 * hl), ld8(enc_h + ((size_t)b * NP + n) * EE + kc * 32 + 8 * hl + 16));
      acc = wmma16(cA[kc], be, acc);
    }
#pragma unroll
    for (int r = 0; r < 8; ++r) {
      int prow = m0 + r + 8 * hl;
      float v = 0.f;
      if (n < NN) {
        float t = 10.0f * tanhf(acc[r] * inv_sqrt_e);   // bounded by CLIP=10
        if (prow < PP) t += mask[((size_t)b * PP + prow) * NN + n];
        v = __expf(t);                                   // no max pass needed
        rsum[r] += v;
      }
      ex[r + 8 * hl][n] = v;
    }
  }
#pragma unroll
  for (int r = 0; r < 8; ++r) rsum[r] = red_sum16(rsum[r]);
  if (lm == 0) {
#pragma unroll
    for (int r = 0; r < 8; ++r) rinv[r + 8 * hl] = 1.0f / rsum[r];
  }
  __syncthreads();

  // vectorized float4 write-out (NN % 4 == 0, all strides 16B aligned)
  for (int row = 0; row < 16; ++row) {
    int p = m0 + row;
    if (p >= PP) break;                      // uniform, rows ascending
    float inv = rinv[row];
    int col0 = lane * 16;
#pragma unroll
    for (int j = 0; j < 4; ++j) {
      int col = col0 + j * 4;
      if (col + 4 <= NN) {
        float4 vv = *(const float4*)&ex[row][col];
        vv.x *= inv; vv.y *= inv; vv.z *= inv; vv.w *= inv;
        *(float4*)&out[((size_t)b * PP + p) * NN + col] = vv;
      }
    }
  }
}

// ---------------- launch ----------------
extern "C" void kernel_launch(void* const* d_in, const int* in_sizes, int n_in,
                              void* d_out, int out_size, void* d_ws, size_t ws_size,
                              hipStream_t stream) {
  (void)in_sizes; (void)n_in; (void)out_size; (void)ws_size;
  const float* enc   = (const float*)d_in[0];
  const float* mask  = (const float*)d_in[1];
  const float* wqf   = (const float*)d_in[2];
  const float* wql   = (const float*)d_in[3];
  const float* wk    = (const float*)d_in[4];
  const float* wv    = (const float*)d_in[5];
  const float* wcomb = (const float*)d_in[6];
  const float* bcomb = (const float*)d_in[7];
  const int*   cur   = (const int*)d_in[8];
  float* out = (float*)d_out;

  _Float16* wsp = (_Float16*)d_ws;
  _Float16* wqf_h = wsp;
  _Float16* wql_h = wqf_h + EE * HD;
  _Float16* wk_h  = wql_h + EE * HD;
  _Float16* wv_h  = wk_h + EE * HD;
  _Float16* wc_h  = wv_h + EE * HD;
  _Float16* enc_h = wc_h + EE * HD;
  const size_t big = (size_t)BB * NP * EE;   // == B*H*NP*D == 4194304 elements
  _Float16* qh     = enc_h + big;
  _Float16* kh     = qh + big;
  _Float16* vt     = kh + big;               // V stored transposed (B,H,D,NP)
  _Float16* attn_h = vt + big;               // total ws use ~42 MB (f16)

  prep_weights<<<(5 * EE * HD + 255) / 256, 256, 0, stream>>>(wqf, wql, wk, wv, wcomb, wsp);
  prep_enc<<<(int)((big + 255) / 256), 256, 0, stream>>>(enc, enc_h);
  qkv_kernel<<<BB * 32, 32, 0, stream>>>(enc_h, wqf_h, wql_h, wk_h, wv_h, cur, qh, kh, vt);
  attn_kernel<<<BB * HH * 32, 32, 0, stream>>>(qh, kh, vt, mask, attn_h);
  final_kernel<<<BB * 32, 32, 0, stream>>>(attn_h, wc_h, bcomb, enc_h, mask, out);
}
